// TransLayer_566935683123
// MI455X (gfx1250) — matrix operand order repaired and checked by reading, not compiled, not run on hardware
//
#include <hip/hip_runtime.h>
#include <hip/hip_bf16.h>
#include <stdint.h>

// ---------------------------------------------------------------------------
// Nystromformer-style layer for MI455X (gfx1250, wave32, WMMA).
// f16 WMMA (16x16x32) for all large GEMMs; f32 WMMA (16x16x4) for the pinv
// Newton-Schulz iteration; async global->LDS staging for the attn3 V tiles.
// One wave per block; A/B fragments use contiguous k order, k = 16*(lane/16) + s.
// ---------------------------------------------------------------------------

typedef _Float16 h16;
typedef __attribute__((ext_vector_type(16))) _Float16 v16h;
typedef __attribute__((ext_vector_type(8)))  float    v8f;
typedef __attribute__((ext_vector_type(2)))  float    v2f;
typedef __attribute__((__vector_size__(4 * sizeof(int)))) int v4i_t;

#define C_H    8
#define C_DH   64
#define C_M    256
#define C_DIM  512
#define C_N    8192
#define C_B    4
#define C_BH   32       // B*H
#define C_NT   32768    // B*N
#define C_K    33

#if defined(__has_builtin)
#  if __has_builtin(__builtin_amdgcn_wmma_f32_16x16x4_f32)
#    define HAVE_WMMA_F32 1
#  endif
#  if __has_builtin(__builtin_amdgcn_global_load_async_to_lds_b128)
#    define HAVE_ASYNC_LDS 1
#  endif
#  if __has_builtin(__builtin_amdgcn_s_wait_asynccnt)
#    define HAVE_WAIT_ASYNC 1
#  endif
#endif
#ifndef HAVE_WMMA_F32
#  define HAVE_WMMA_F32 0
#endif
#ifndef HAVE_ASYNC_LDS
#  define HAVE_ASYNC_LDS 0
#endif
#ifndef HAVE_WAIT_ASYNC
#  define HAVE_WAIT_ASYNC 0
#endif

// ---- WMMA fragment helpers (wave32; A/B in contiguous k order k = 16*(lane/16) + s) --------------
static __device__ __forceinline__ v16h load_a16(const h16* A, int lda, int ks, int lane) {
  int row = lane & 15, lh = lane >> 4;
  const unsigned int* p = (const unsigned int*)(A + row * lda + ks);
  union { v16h v; unsigned int u[8]; } f;
#pragma unroll
  for (int j = 0; j < 8; ++j) f.u[j] = p[lh * 8 + j];
  return f.v;
}
static __device__ __forceinline__ v16h load_b16(const h16* B, int ldb, int ks, int lane) {
  int n = lane & 15, lh = lane >> 4;
  const unsigned int* p = (const unsigned int*)(B + n * ldb + ks);
  union { v16h v; unsigned int u[8]; } f;
#pragma unroll
  for (int j = 0; j < 8; ++j) f.u[j] = p[lh * 8 + j];
  return f.v;
}
static __device__ __forceinline__ v8f wmma_f16(v16h a, v16h b, v8f c) {
  return __builtin_amdgcn_wmma_f32_16x16x32_f16(false, a, false, b, (short)0, c, false, false);
}

// ---- 1. LayerNorm: x fp32 [NT,512] -> xn f16 -------------------------------
__global__ void ln_kernel(const float* __restrict__ x, const float* __restrict__ w,
                          const float* __restrict__ bsh, h16* __restrict__ xn) {
  int row = blockIdx.x, lane = threadIdx.x;
  const float* xr = x + (size_t)row * C_DIM;
  float v[16], s = 0.f;
#pragma unroll
  for (int j = 0; j < 16; ++j) { v[j] = xr[lane + 32 * j]; s += v[j]; }
#pragma unroll
  for (int off = 16; off > 0; off >>= 1) s += __shfl_xor(s, off, 32);
  float mu = s * (1.f / C_DIM), q = 0.f;
#pragma unroll
  for (int j = 0; j < 16; ++j) { float d = v[j] - mu; q += d * d; }
#pragma unroll
  for (int off = 16; off > 0; off >>= 1) q += __shfl_xor(q, off, 32);
  float rstd = rsqrtf(q * (1.f / C_DIM) + 1e-5f);
#pragma unroll
  for (int j = 0; j < 16; ++j) {
    int c = lane + 32 * j;
    xn[(size_t)row * C_DIM + c] = (h16)((v[j] - mu) * rstd * w[c] + bsh[c]);
  }
}

// ---- 2. fp32 -> f16 convert ------------------------------------------------
__global__ void cvt_kernel(const float* __restrict__ src, h16* __restrict__ dst, int n) {
  int i = blockIdx.x * 256 + threadIdx.x;
  if (i < n) dst[i] = (h16)src[i];
}

// ---- 3. QKV GEMM: xn[NT,512] @ w_qkv[1536,512]^T (pipelined), scatter heads
__global__ void gemm_qkv_kernel(const h16* __restrict__ xn, const h16* __restrict__ wq,
                                h16* __restrict__ qh, h16* __restrict__ kh, h16* __restrict__ vh) {
  int lane = threadIdx.x, lh = lane >> 4, cc = lane & 15;
  int rowbase = blockIdx.y * 16, colbase = blockIdx.x * 64;
  const h16* Ab = xn + (size_t)rowbase * C_DIM;
  v8f c4[4] = {};
  v16h a_c = load_a16(Ab, C_DIM, 0, lane);
  v16h b_c[4], b_n[4];
#pragma unroll
  for (int nt = 0; nt < 4; ++nt)
    b_c[nt] = load_b16(wq + (size_t)(colbase + nt * 16) * C_DIM, C_DIM, 0, lane);
  for (int ks = 32; ks < C_DIM; ks += 32) {
    v16h a_n = load_a16(Ab, C_DIM, ks, lane);
#pragma unroll
    for (int nt = 0; nt < 4; ++nt)
      b_n[nt] = load_b16(wq + (size_t)(colbase + nt * 16) * C_DIM, C_DIM, ks, lane);
#pragma unroll
    for (int nt = 0; nt < 4; ++nt) c4[nt] = wmma_f16(a_c, b_c[nt], c4[nt]);
    a_c = a_n;
#pragma unroll
    for (int nt = 0; nt < 4; ++nt) b_c[nt] = b_n[nt];
  }
#pragma unroll
  for (int nt = 0; nt < 4; ++nt) c4[nt] = wmma_f16(a_c, b_c[nt], c4[nt]);

  int which = colbase / C_DIM;                 // 0=q 1=k 2=v
  int h = (colbase % C_DIM) / C_DH;
  h16* dst = which == 0 ? qh : (which == 1 ? kh : vh);
  float sc = which == 0 ? 0.125f : 1.f;        // DH^-0.5
#pragma unroll
  for (int nt = 0; nt < 4; ++nt)
#pragma unroll
    for (int r = 0; r < 8; ++r) {
      int i = rowbase + r + 8 * lh, b = i >> 13, t = i & 8191, d = nt * 16 + cc;
      dst[(((size_t)(b * C_H + h)) * C_N + t) * C_DH + d] = (h16)(c4[nt][r] * sc);
    }
}

// ---- 4. Landmark mean-pool: [bh,8192,64] -> [bh,256,64] --------------------
__global__ void pool_kernel(const h16* __restrict__ src, h16* __restrict__ dst) {
  int idx = blockIdx.x * 256 + threadIdx.x;     // 32*256*64
  int bh = idx >> 14, rem = idx & 16383, m = rem >> 6, d = rem & 63;
  const h16* sp = src + (size_t)bh * C_N * C_DH + (size_t)m * 32 * C_DH + d;
  float s = 0.f;
#pragma unroll 8
  for (int u = 0; u < 32; ++u) s += (float)sp[u * C_DH];
  dst[idx] = (h16)(s * (1.f / 32.f));
}

// ---- 5. Fused  softmax(Q @ Kl^T)  [+ optional  @ W2 epilogue] --------------
// mode 0: write normalized probs fp32 (attn2).   mode 1: O1 = P @ W2 (attn1 path).
__global__ void attn_sm_kernel(const h16* __restrict__ Q, const h16* __restrict__ Kl,
                               const h16* __restrict__ W2T, float* __restrict__ A2out,
                               float* __restrict__ O1, int mode, int rowsPerBH) {
  __shared__ float Sf[16][256];
  __shared__ h16   Pb[16][256];
  int bh = blockIdx.y, rowbase = blockIdx.x * 16;
  int lane = threadIdx.x, lh = lane >> 4, cc = lane & 15;
  const h16* Qb = Q + (size_t)bh * rowsPerBH * C_DH + (size_t)rowbase * C_DH;
  const h16* Kb = Kl + (size_t)bh * C_M * C_DH;
  // Q fragments are invariant across all 16 column tiles: hoist.
  v16h aq0 = load_a16(Qb, C_DH, 0, lane);
  v16h aq1 = load_a16(Qb, C_DH, 32, lane);
  for (int nt = 0; nt < 16; ++nt) {
    v8f c = {};
    v16h b0 = load_b16(Kb + nt * 16 * C_DH, C_DH, 0, lane);
    v16h b1 = load_b16(Kb + nt * 16 * C_DH, C_DH, 32, lane);
    c = wmma_f16(aq0, b0, c);
    c = wmma_f16(aq1, b1, c);
#pragma unroll
    for (int r = 0; r < 8; ++r) Sf[r + 8 * lh][nt * 16 + cc] = c[r];
  }
  __syncthreads();
  // 32-lane softmax: lane-halves split each row, combine via shfl_xor(16).
  {
    int row = cc, base = lh * 128;
    float m = -1e30f;
    for (int u = 0; u < 128; ++u) m = fmaxf(m, Sf[row][base + u]);
    m = fmaxf(m, __shfl_xor(m, 16, 32));
    float s = 0.f;
    for (int u = 0; u < 128; ++u) s += __expf(Sf[row][base + u] - m);
    s += __shfl_xor(s, 16, 32);
    float inv = 1.f / s;
    for (int u = 0; u < 128; ++u) {
      float p = __expf(Sf[row][base + u] - m) * inv;
      if (mode == 0) Sf[row][base + u] = p; else Pb[row][base + u] = (h16)p;
    }
  }
  __syncthreads();
  if (mode == 0) {
    float* dst = A2out + (size_t)bh * C_M * C_M + (size_t)rowbase * C_M;
    for (int idx = lane; idx < 4096; idx += 32) dst[idx] = Sf[idx >> 8][idx & 255];
  } else {
    const h16* Wb = W2T + (size_t)bh * C_DH * C_M;    // W2T[d][m], ld=256
    v8f c4[4] = {};
    for (int ks = 0; ks < C_M; ks += 32) {
      v16h a = load_a16(&Pb[0][0], 256, ks, lane);
#pragma unroll
      for (int nt = 0; nt < 4; ++nt) {
        v16h b = load_b16(Wb + nt * 16 * C_M, C_M, ks, lane);
        c4[nt] = wmma_f16(a, b, c4[nt]);
      }
    }
    float* Ob = O1 + (size_t)bh * rowsPerBH * C_DH + (size_t)rowbase * C_DH;
#pragma unroll
    for (int nt = 0; nt < 4; ++nt)
#pragma unroll
      for (int r = 0; r < 8; ++r) Ob[(r + 8 * lh) * C_DH + nt * 16 + cc] = c4[nt][r];
  }
}

// ---- 6. pinv init reductions ----------------------------------------------
__global__ void rowcol_kernel(const float* __restrict__ a2, float* __restrict__ rs,
                              float* __restrict__ cs) {
  int t = blockIdx.x * 256 + threadIdx.x;     // 16384 threads
  if (t < 8192) {
    int bh = t >> 8, j = t & 255; float s = 0.f;
    for (int i = 0; i < 256; ++i) s += a2[(size_t)bh * 65536 + i * 256 + j];
    cs[t] = s;
  } else {
    int u = t - 8192, bh = u >> 8, i = u & 255; float s = 0.f;
    for (int j = 0; j < 256; ++j) s += a2[(size_t)bh * 65536 + i * 256 + j];
    rs[u] = s;
  }
}
__global__ void maxmul_kernel(const float* __restrict__ rs, const float* __restrict__ cs,
                              float* __restrict__ scal) {
  __shared__ float sa[256], sb[256];
  int t = threadIdx.x; float mr = -1e30f, mc = -1e30f;
  for (int i = t; i < 8192; i += 256) { mr = fmaxf(mr, rs[i]); mc = fmaxf(mc, cs[i]); }
  sa[t] = mr; sb[t] = mc; __syncthreads();
  for (int s = 128; s > 0; s >>= 1) {
    if (t < s) { sa[t] = fmaxf(sa[t], sa[t + s]); sb[t] = fmaxf(sb[t], sb[t + s]); }
    __syncthreads();
  }
  if (t == 0) scal[0] = 1.f / (sa[0] * sb[0]);
}
__global__ void zinit_kernel(const float* __restrict__ a2, const float* __restrict__ scal,
                             float* __restrict__ z0) {
  int idx = blockIdx.x * 256 + threadIdx.x;   // 32*256*256
  int bh = idx >> 16, r = (idx >> 8) & 255, c = idx & 255;
  z0[idx] = a2[((size_t)bh << 16) + c * 256 + r] * scal[0];
}

// ---- 7. 256x256x256 fp32 matmul:  C = scale * A @ (useI? cval*I - B : B) ---
__global__ void mm256_kernel(float* __restrict__ C, const float* __restrict__ A,
                             const float* __restrict__ B, int useI, float cval, float scale) {
  int bh = blockIdx.y, tile = blockIdx.x;
  int ri = (tile >> 4) * 16, cj = (tile & 15) * 16;
  int lane = threadIdx.x, lh = lane >> 4, col = lane & 15;
  const float* Ab = A + (size_t)bh * 65536;
  const float* Bb = B + (size_t)bh * 65536;
  float*       Cb = C + (size_t)bh * 65536;
#if HAVE_WMMA_F32
  v8f acc = {};
  for (int kb = 0; kb < 256; kb += 4) {
    v2f a, b;
    a[0] = Ab[(ri + col) * 256 + kb + 2 * lh];
    a[1] = Ab[(ri + col) * 256 + kb + 2 * lh + 1];
    int k0 = kb + 2 * lh;
    float b0 = Bb[(size_t)k0 * 256 + cj + col];
    float b1 = Bb[(size_t)(k0 + 1) * 256 + cj + col];
    if (useI) {
      b0 = ((k0 == cj + col) ? cval : 0.f) - b0;
      b1 = ((k0 + 1 == cj + col) ? cval : 0.f) - b1;
    }
    b[0] = b0; b[1] = b1;
    acc = __builtin_amdgcn_wmma_f32_16x16x4_f32(false, a, false, b, (short)0, acc, false, false);
  }
#pragma unroll
  for (int r = 0; r < 8; ++r) Cb[(ri + r + 8 * lh) * 256 + cj + col] = scale * acc[r];
#else
  float acc[8] = {0, 0, 0, 0, 0, 0, 0, 0};
  for (int kb = 0; kb < 256; ++kb) {
    float bv = Bb[(size_t)kb * 256 + cj + col];
    if (useI) bv = ((kb == cj + col) ? cval : 0.f) - bv;
#pragma unroll
    for (int r = 0; r < 8; ++r) acc[r] += Ab[(ri + r + 8 * lh) * 256 + kb] * bv;
  }
#pragma unroll
  for (int r = 0; r < 8; ++r) Cb[(ri + r + 8 * lh) * 256 + cj + col] = scale * acc[r];
#endif
}

// ---- 8. kv = softmax(q_l @ k^T) @ v  (flash-style two-pass over n=8192) ----
__global__ void attn3_kv_kernel(const h16* __restrict__ ql, const h16* __restrict__ kh,
                                const h16* __restrict__ vh, float* __restrict__ kv) {
  __shared__ float St[16][32];
  __shared__ h16   Pb2[16][32];
  __shared__ h16   Vt[32][64];                 // staged V tile (async -> LDS)
  __shared__ float mrow[16], sinv[16];
  int bh = blockIdx.y, rowbase = blockIdx.x * 16;
  int lane = threadIdx.x, lh = lane >> 4, cc = lane & 15;
  const h16* Qb = ql + (size_t)bh * C_M * C_DH + (size_t)rowbase * C_DH;
  const h16* Kb = kh + (size_t)bh * C_N * C_DH;
  const h16* Vb = vh + (size_t)bh * C_N * C_DH;
  // Q fragments invariant over the whole key stream: load once.
  v16h aq0 = load_a16(Qb, C_DH, 0, lane);
  v16h aq1 = load_a16(Qb, C_DH, 32, lane);
  float m = -1e30f, s = 0.f;
  // ---- pass 1: online row max & sumexp ----
  for (int jt = 0; jt < 256; ++jt) {
#pragma unroll
    for (int sub = 0; sub < 2; ++sub) {
      v8f c = {};
      v16h b0 = load_b16(Kb + (size_t)(jt * 32 + sub * 16) * C_DH, C_DH, 0, lane);
      v16h b1 = load_b16(Kb + (size_t)(jt * 32 + sub * 16) * C_DH, C_DH, 32, lane);
      c = wmma_f16(aq0, b0, c);
      c = wmma_f16(aq1, b1, c);
#pragma unroll
      for (int r = 0; r < 8; ++r) St[r + 8 * lh][sub * 16 + cc] = c[r];
    }
    __syncthreads();
    {
      int row = cc, base = lh * 16;
      float lm = -1e30f;
#pragma unroll
      for (int u = 0; u < 16; ++u) lm = fmaxf(lm, St[row][base + u]);
      lm = fmaxf(lm, __shfl_xor(lm, 16, 32));
      float mn = fmaxf(m, lm), ss = 0.f;
#pragma unroll
      for (int u = 0; u < 16; ++u) ss += __expf(St[row][base + u] - mn);
      ss += __shfl_xor(ss, 16, 32);
      s = s * __expf(m - mn) + ss; m = mn;
    }
    __syncthreads();
  }
  if (lane < 16) { mrow[lane] = m; sinv[lane] = 1.f / s; }
  __syncthreads();
  // ---- pass 2: recompute, normalize, accumulate  P @ V ----
  v8f c4[4] = {};
  for (int jt = 0; jt < 256; ++jt) {
    // stage V tile [32 keys x 64 dims] into LDS (async DMA path on gfx1250)
    {
      const h16* gp = Vb + (size_t)(jt * 32 + lane) * C_DH;
      h16* lrow = &Vt[lane][0];
#if HAVE_ASYNC_LDS
#pragma unroll
      for (int i2 = 0; i2 < 8; ++i2)
        __builtin_amdgcn_global_load_async_to_lds_b128(
            (__attribute__((address_space(1))) v4i_t*)(gp + i2 * 8),
            (__attribute__((address_space(3))) v4i_t*)(lrow + i2 * 8), 0, 0);
#  if HAVE_WAIT_ASYNC
      __builtin_amdgcn_s_wait_asynccnt(0);
#  else
      asm volatile("s_wait_asynccnt 0x0" ::: "memory");
#  endif
#else
      const uint4* g4 = (const uint4*)gp;
      uint4* l4 = (uint4*)lrow;
#pragma unroll
      for (int i2 = 0; i2 < 8; ++i2) l4[i2] = g4[i2];
#endif
    }
#pragma unroll
    for (int sub = 0; sub < 2; ++sub) {
      v8f c = {};
      v16h b0 = load_b16(Kb + (size_t)(jt * 32 + sub * 16) * C_DH, C_DH, 0, lane);
      v16h b1 = load_b16(Kb + (size_t)(jt * 32 + sub * 16) * C_DH, C_DH, 32, lane);
      c = wmma_f16(aq0, b0, c);
      c = wmma_f16(aq1, b1, c);
#pragma unroll
      for (int r = 0; r < 8; ++r) St[r + 8 * lh][sub * 16 + cc] = c[r];
    }
    __syncthreads();
    {
      int row = cc, base = lh * 16;
#pragma unroll
      for (int u = 0; u < 16; ++u)
        Pb2[row][base + u] = (h16)(__expf(St[row][base + u] - mrow[row]) * sinv[row]);
    }
    __syncthreads();
    v16h a = load_a16(&Pb2[0][0], 32, 0, lane);
#pragma unroll
    for (int nt = 0; nt < 4; ++nt) {
      union { v16h v; h16 h[16]; } bf;
      int dim = nt * 16 + cc;
#pragma unroll
      for (int e = 0; e < 16; ++e) bf.h[e] = Vt[lh * 16 + e][dim];
      c4[nt] = wmma_f16(a, bf.v, c4[nt]);
    }
    __syncthreads();
  }
  float* Kvb = kv + (size_t)bh * C_M * C_DH + (size_t)rowbase * C_DH;
#pragma unroll
  for (int nt = 0; nt < 4; ++nt)
#pragma unroll
    for (int r = 0; r < 8; ++r) Kvb[(r + 8 * lh) * C_DH + nt * 16 + cc] = c4[nt][r];
}

// ---- 9. W2T[d][m] = (Z^-1 @ kv)^T  (tiny, plain FMA) -----------------------
__global__ void w2t_kernel(const float* __restrict__ z, const float* __restrict__ kv,
                           h16* __restrict__ w2t) {
  int idx = blockIdx.x * 256 + threadIdx.x;   // 32*64*256
  int bh = idx >> 14, rem = idx & 16383, d = rem >> 8, i = rem & 255;
  const float* Zb = z + (size_t)bh * 65536 + (size_t)i * 256;
  const float* Kb = kv + (size_t)bh * C_M * C_DH + d;
  float s = 0.f;
  for (int j = 0; j < 256; ++j) s += Zb[j] * Kb[j * C_DH];
  w2t[(size_t)bh * C_DH * C_M + d * 256 + i] = (h16)s;
}

// ---- 10. depthwise conv over seq + head re-assembly ------------------------
__global__ void conv_assemble_kernel(const float* __restrict__ o1, const h16* __restrict__ vh,
                                     const float* __restrict__ rk, h16* __restrict__ yb) {
  int idx = blockIdx.x * 256 + threadIdx.x;   // NT*DIM
  int i = idx >> 9, c = idx & 511, h = c >> 6, d = c & 63;
  int b = i >> 13, t = i & 8191, bh = b * C_H + h;
  size_t basev = (size_t)bh * C_N * C_DH + d;
  float acc = o1[((size_t)bh * C_N + t) * C_DH + d];
#pragma unroll
  for (int u = 0; u < C_K; ++u) {
    int tt = t + u - (C_K / 2);
    if (tt >= 0 && tt < C_N) acc += rk[h * C_K + u] * (float)vh[basev + (size_t)tt * C_DH];
  }
  yb[idx] = (h16)acc;
}

// ---- 11. Output GEMM: y[NT,512] @ w_out^T + b_out + x (pipelined) ----------
__global__ void gemm_out_kernel(const h16* __restrict__ yb, const h16* __restrict__ wo,
                                const float* __restrict__ bo, const float* __restrict__ x,
                                float* __restrict__ out) {
  int lane = threadIdx.x, lh = lane >> 4, cc = lane & 15;
  int rowbase = blockIdx.y * 16, colbase = blockIdx.x * 64;
  const h16* Ab = yb + (size_t)rowbase * C_DIM;
  v8f c4[4] = {};
  v16h a_c = load_a16(Ab, C_DIM, 0, lane);
  v16h b_c[4], b_n[4];
#pragma unroll
  for (int nt = 0; nt < 4; ++nt)
    b_c[nt] = load_b16(wo + (size_t)(colbase + nt * 16) * C_DIM, C_DIM, 0, lane);
  for (int ks = 32; ks < C_DIM; ks += 32) {
    v16h a_n = load_a16(Ab, C_DIM, ks, lane);
#pragma unroll
    for (int nt = 0; nt < 4; ++nt)
      b_n[nt] = load_b16(wo + (size_t)(colbase + nt * 16) * C_DIM, C_DIM, ks, lane);
#pragma unroll
    for (int nt = 0; nt < 4; ++nt) c4[nt] = wmma_f16(a_c, b_c[nt], c4[nt]);
    a_c = a_n;
#pragma unroll
    for (int nt = 0; nt < 4; ++nt) b_c[nt] = b_n[nt];
  }
#pragma unroll
  for (int nt = 0; nt < 4; ++nt) c4[nt] = wmma_f16(a_c, b_c[nt], c4[nt]);
#pragma unroll
  for (int nt = 0; nt < 4; ++nt)
#pragma unroll
    for (int r = 0; r < 8; ++r) {
      int i = rowbase + r + 8 * lh, j = colbase + nt * 16 + cc;
      out[(size_t)i * C_DIM + j] = c4[nt][r] + bo[j] + x[(size_t)i * C_DIM + j];
    }
}

// ---------------------------------------------------------------------------
extern "C" void kernel_launch(void* const* d_in, const int* in_sizes, int n_in,
                              void* d_out, int out_size, void* d_ws, size_t ws_size,
                              hipStream_t stream) {
  const float* x    = (const float*)d_in[0];
  const float* ln_w = (const float*)d_in[1];
  const float* ln_b = (const float*)d_in[2];
  const float* wqkv = (const float*)d_in[3];
  const float* wout = (const float*)d_in[4];
  const float* bout = (const float*)d_in[5];
  const float* rk   = (const float*)d_in[6];
  float* out = (float*)d_out;

  size_t off = 0;
  auto alloc = [&](size_t bytes) -> char* {
    char* p = (char*)d_ws + off; off += (bytes + 255) & ~(size_t)255; return p;
  };
  h16*   xn_h = (h16*)  alloc((size_t)C_NT * C_DIM * 2);
  h16*   qh   = (h16*)  alloc((size_t)C_NT * C_DIM * 2);
  h16*   kh   = (h16*)  alloc((size_t)C_NT * C_DIM * 2);
  h16*   vh   = (h16*)  alloc((size_t)C_NT * C_DIM * 2);
  h16*   ql   = (h16*)  alloc((size_t)C_BH * C_M * C_DH * 2);
  h16*   kl   = (h16*)  alloc((size_t)C_BH * C_M * C_DH * 2);
  h16*   wq_h = (h16*)  alloc((size_t)3 * C_DIM * C_DIM * 2);
  h16*   wo_h = (h16*)  alloc((size_t)C_DIM * C_DIM * 2);
  float* a2   = (float*)alloc((size_t)C_BH * C_M * C_M * 4);
  float* z0   = (float*)alloc((size_t)C_BH * C_M * C_M * 4);
  float* z1   = (float*)alloc((size_t)C_BH * C_M * C_M * 4);
  float* xz   = (float*)alloc((size_t)C_BH * C_M * C_M * 4);
  float* t1   = (float*)alloc((size_t)C_BH * C_M * C_M * 4);
  float* t2   = (float*)alloc((size_t)C_BH * C_M * C_M * 4);
  float* kv   = (float*)alloc((size_t)C_BH * C_M * C_DH * 4);
  h16*   w2t  = (h16*)  alloc((size_t)C_BH * C_DH * C_M * 2);
  float* o1   = (float*)alloc((size_t)C_NT * C_DIM * 4);
  h16*   yb   = (h16*)  alloc((size_t)C_NT * C_DIM * 2);
  float* rsum = (float*)alloc((size_t)C_BH * C_M * 4);
  float* csum = (float*)alloc((size_t)C_BH * C_M * 4);
  float* scal = (float*)alloc(256);

  cvt_kernel<<<(3 * C_DIM * C_DIM + 255) / 256, 256, 0, stream>>>(wqkv, wq_h, 3 * C_DIM * C_DIM);
  cvt_kernel<<<(C_DIM * C_DIM + 255) / 256, 256, 0, stream>>>(wout, wo_h, C_DIM * C_DIM);
  ln_kernel<<<C_NT, 32, 0, stream>>>(x, ln_w, ln_b, xn_h);
  gemm_qkv_kernel<<<dim3(3 * C_DIM / 64, C_NT / 16), 32, 0, stream>>>(xn_h, wq_h, qh, kh, vh);
  pool_kernel<<<(C_BH * C_M * C_DH) / 256, 256, 0, stream>>>(qh, ql);
  pool_kernel<<<(C_BH * C_M * C_DH) / 256, 256, 0, stream>>>(kh, kl);
  attn_sm_kernel<<<dim3(C_M / 16, C_BH), 32, 0, stream>>>(ql, kl, nullptr, a2, nullptr, 0, C_M);
  rowcol_kernel<<<64, 256, 0, stream>>>(a2, rsum, csum);
  maxmul_kernel<<<1, 256, 0, stream>>>(rsum, csum, scal);
  zinit_kernel<<<(C_BH * C_M * C_M) / 256, 256, 0, stream>>>(a2, scal, z0);
  float* zc = z0; float* zn = z1;
  for (int it = 0; it < 6; ++it) {
    mm256_kernel<<<dim3(256, C_BH), 32, 0, stream>>>(xz, a2, zc, 0, 0.f, 1.f);
    mm256_kernel<<<dim3(256, C_BH), 32, 0, stream>>>(t1, xz, xz, 1, 7.f, 1.f);
    mm256_kernel<<<dim3(256, C_BH), 32, 0, stream>>>(t2, xz, t1, 1, 15.f, 1.f);
    mm256_kernel<<<dim3(256, C_BH), 32, 0, stream>>>(zn, zc, t2, 1, 13.f, 0.25f);
    float* tmp = zc; zc = zn; zn = tmp;
  }
  attn3_kv_kernel<<<dim3(C_M / 16, C_BH), 32, 0, stream>>>(ql, kh, vh, kv);
  w2t_kernel<<<(C_BH * C_DH * C_M) / 256, 256, 0, stream>>>(zc, kv, w2t);
  attn_sm_kernel<<<dim3(C_N / 16, C_BH), 32, 0, stream>>>(qh, kl, w2t, nullptr, o1, 1, C_N);
  conv_assemble_kernel<<<(C_NT * C_DIM) / 256, 256, 0, stream>>>(o1, vh, rk, yb);
  gemm_out_kernel<<<dim3(C_DIM / 64, C_NT / 16), 32, 0, stream>>>(yb, wo_h, bout, x, out);
  (void)in_sizes; (void)n_in; (void)out_size; (void)ws_size;
}
